// GnnCritic_14276471292239
// MI455X (gfx1250) — matrix operand order repaired and checked by reading, not compiled, not run on hardware
//
#include <hip/hip_runtime.h>

// ---------- types ----------
typedef __attribute__((ext_vector_type(16))) __bf16 v16bf;
typedef __attribute__((ext_vector_type(8)))  float  v8f;
typedef unsigned short u16_t;
typedef __attribute__((ext_vector_type(8))) unsigned short u16x8;

union BPack { u16x8 h[2]; v16bf v; };

// ---------- problem constants ----------
#define BODY 10
#define OBJD 15
#define OBSD 55          // 10 + 3*15
#define D_MP 128
#define KP1  160         // 153 padded to 160
#define HID  256
#define HSTR 264         // hidden LDS row stride (bf16 elems), 528B = 16B aligned
#define MROWS 32         // batch rows per block

// weight layout in d_ws (u16 element offsets)
#define OFF_W1 0
#define OFF_W3 (256*160)
#define OFF_W2 (2*256*160)
#define OFF_W4 (2*256*160 + 256*256)
#define WS_TOT (2*256*160 + 2*256*256)   // 212992 u16 = 416 KB

__device__ __forceinline__ unsigned short f2bf(float f) {
    union { float f; unsigned u; } x; x.f = f;
    unsigned u = x.u;
    u += 0x7FFFu + ((u >> 16) & 1u);     // round-to-nearest-even
    return (unsigned short)(u >> 16);
}

// A-operand (16xK tile, row-major bf16 in LDS): lane m = lane&15,
// elems 0-7 -> K = kb + half*8 + j ; elems 8-15 -> K = kb + 16 + half*8 + j
__device__ __forceinline__ v16bf load_a(const u16_t* base, int stride,
                                        int l15, int half, int kb) {
    const u16_t* row = base + l15 * stride + kb + half * 8;
    BPack p;
    p.h[0] = *(const u16x8*)(row);
    p.h[1] = *(const u16x8*)(row + 16);
    return p.v;
}

// B-operand (Kx16 tile) from weights stored transposed [N][Kpad] bf16:
// lane column n = nt*16 + (lane&15); elems 0-7 -> K = kb + half*8 + j ; elems 8-15 -> K = kb + 16 + half*8 + j
__device__ __forceinline__ v16bf load_b(const u16_t* wt, int stride,
                                        int l15, int half, int kb, int nt) {
    const u16_t* row = wt + (nt * 16 + l15) * stride + kb + half * 8;
    BPack p;
    p.h[0] = *(const u16x8*)(row);
    p.h[1] = *(const u16x8*)(row + 16);
    return p.v;
}

// ---------- weight conversion: fp32 [K][N] -> bf16 transposed [N][Kpad] ----------
__global__ __launch_bounds__(256) void conv_weights(
    const float* __restrict__ w1, const float* __restrict__ w2,
    const float* __restrict__ w3, const float* __restrict__ w4,
    u16_t* __restrict__ ws) {
    int idx = blockIdx.x * 256 + threadIdx.x;
    if (idx < OFF_W3) {                       // w1t: 256 x 160
        int n = idx / 160, k = idx % 160;
        float v = (k < 153) ? w1[k * 256 + n] : 0.0f;
        ws[OFF_W1 + idx] = f2bf(v);
    } else if (idx < OFF_W2) {                // w3t
        int j = idx - OFF_W3;
        int n = j / 160, k = j % 160;
        float v = (k < 153) ? w3[k * 256 + n] : 0.0f;
        ws[idx] = f2bf(v);
    } else if (idx < OFF_W4) {                // w2t: 256 x 256
        int j = idx - OFF_W2;
        int n = j / 256, k = j % 256;
        ws[idx] = f2bf(w2[k * 256 + n]);
    } else if (idx < WS_TOT) {                // w4t
        int j = idx - OFF_W4;
        int n = j / 256, k = j % 256;
        ws[idx] = f2bf(w4[k * 256 + n]);
    }
}

// ---------- fused GNN critic ----------
__global__ __launch_bounds__(256) void gnn_critic_main(
    const float* __restrict__ obs, const float* __restrict__ ef,
    const int*   __restrict__ edges_to,
    const float* __restrict__ b1, const float* __restrict__ b2,
    const float* __restrict__ b3, const float* __restrict__ b4,
    const float* __restrict__ wr1, const float* __restrict__ br1,
    const float* __restrict__ wr2, const float* __restrict__ br2,
    const u16_t* __restrict__ ws, float* __restrict__ out, int Btot) {

    __shared__ u16_t a_lds[3][MROWS][KP1];    // 30720 B
    __shared__ u16_t h_lds[MROWS][HSTR];      // 16896 B
    __shared__ float bias1[2][HID];
    __shared__ float bias2[2][HID];
    __shared__ float wr_s[2][HID];
    __shared__ float qacc[2][MROWS];
    __shared__ int   e_list[3][6];
    __shared__ int   e_cnt[3];

    const int tid  = threadIdx.x;
    const int b0   = blockIdx.x * MROWS;
    const int wave = tid >> 5;
    const int lane = tid & 31;
    const int half = lane >> 4;
    const int l15  = lane & 15;

    // per-node edge lists (segment_max membership)
    if (tid < 3) {
        int c = 0;
        for (int e = 0; e < 6; ++e)
            if (edges_to[e] == tid) e_list[tid][c++] = e;
        e_cnt[tid] = c;
    }
    // biases / readout weights to LDS (tid in [0,256))
    bias1[0][tid] = b1[tid];  bias1[1][tid] = b3[tid];
    bias2[0][tid] = b2[tid];  bias2[1][tid] = b4[tid];
    wr_s[0][tid]  = wr1[tid]; wr_s[1][tid]  = wr2[tid];
    if (tid < 2 * MROWS) qacc[tid >> 5][tid & (MROWS - 1)] = 0.0f;
    __syncthreads();

    // ---- stage body/obj + zero pad (cols 0..24, 153..159) ----
    for (int i = tid; i < 3 * MROWS * 32; i += 256) {
        int n = i / (MROWS * 32);
        int r = (i / 32) % MROWS;
        int c = i % 32;
        int rg = b0 + r; if (rg >= Btot) rg = Btot - 1;
        float v; int col;
        if (c < BODY)      { col = c;             v = obs[(size_t)rg * OBSD + c]; }
        else if (c < 25)   { col = c;             v = obs[(size_t)rg * OBSD + BODY + n * OBJD + (c - BODY)]; }
        else               { col = 153 + (c - 25); v = 0.0f; }
        a_lds[n][r][col] = f2bf(v);
    }
    // ---- stage agg = segment_max(edge_features) (cols 25..152), float4 ----
    for (int i = tid; i < 3 * MROWS * 32; i += 256) {
        int n  = i / (MROWS * 32);
        int r  = (i / 32) % MROWS;
        int c4 = (i % 32) * 4;
        int rg = b0 + r; if (rg >= Btot) rg = Btot - 1;
        float m0 = -3.4e38f, m1 = -3.4e38f, m2 = -3.4e38f, m3 = -3.4e38f;
        int cnt = e_cnt[n];
        for (int j = 0; j < cnt; ++j) {
            const float4 t = *(const float4*)(ef + (((size_t)rg * 6 + e_list[n][j]) * D_MP + c4));
            m0 = fmaxf(m0, t.x); m1 = fmaxf(m1, t.y);
            m2 = fmaxf(m2, t.z); m3 = fmaxf(m3, t.w);
        }
        a_lds[n][r][25 + c4 + 0] = f2bf(m0);
        a_lds[n][r][25 + c4 + 1] = f2bf(m1);
        a_lds[n][r][25 + c4 + 2] = f2bf(m2);
        a_lds[n][r][25 + c4 + 3] = f2bf(m3);
    }
    __syncthreads();

    const v8f zero8 = {0.f, 0.f, 0.f, 0.f, 0.f, 0.f, 0.f, 0.f};
    const int nt0 = wave * 2;                 // this wave's two N-tiles
    float racc[2][8];                         // per-lane readout partials

    for (int combo = 0; combo < 6; ++combo) {
        const int node = combo % 3;
        const int mlp  = combo / 3;
        const u16_t* w_l1 = ws + (mlp ? OFF_W3 : OFF_W1);
        const u16_t* w_l2 = ws + (mlp ? OFF_W4 : OFF_W2);

        if (node == 0) {
            #pragma unroll
            for (int mt = 0; mt < 2; ++mt)
                #pragma unroll
                for (int r = 0; r < 8; ++r) racc[mt][r] = 0.0f;
        }

        // ---- layer 1: hidden = relu(inp @ W + b), bf16 to h_lds ----
        {
            v8f acc[2][2] = {{zero8, zero8}, {zero8, zero8}};   // [nn][mt]
            #pragma unroll
            for (int ks = 0; ks < 5; ++ks) {
                const int kb = ks * 32;
                v16bf am[2], bm[2];
                #pragma unroll
                for (int mt = 0; mt < 2; ++mt)
                    am[mt] = load_a(&a_lds[node][mt * 16][0], KP1, l15, half, kb);
                #pragma unroll
                for (int nn = 0; nn < 2; ++nn)
                    bm[nn] = load_b(w_l1, KP1, l15, half, kb, nt0 + nn);
                #pragma unroll
                for (int nn = 0; nn < 2; ++nn)
                    #pragma unroll
                    for (int mt = 0; mt < 2; ++mt)
                        acc[nn][mt] = __builtin_amdgcn_wmma_f32_16x16x32_bf16(
                            false, am[mt], false, bm[nn], (short)0, acc[nn][mt],
                            false, false);
            }
            #pragma unroll
            for (int nn = 0; nn < 2; ++nn) {
                const int col = (nt0 + nn) * 16 + l15;
                const float bv = bias1[mlp][col];
                #pragma unroll
                for (int mt = 0; mt < 2; ++mt)
                    #pragma unroll
                    for (int r = 0; r < 8; ++r) {
                        float h = fmaxf(acc[nn][mt][r] + bv, 0.0f);
                        h_lds[mt * 16 + half * 8 + r][col] = f2bf(h);
                    }
            }
        }
        __syncthreads();

        // ---- layer 2 + fused 256->1 readout ----
        {
            v8f acc[2][2] = {{zero8, zero8}, {zero8, zero8}};   // [nn][mt]
            #pragma unroll
            for (int ks = 0; ks < 8; ++ks) {
                const int kb = ks * 32;
                v16bf am[2], bm[2];
                #pragma unroll
                for (int mt = 0; mt < 2; ++mt)
                    am[mt] = load_a(&h_lds[mt * 16][0], HSTR, l15, half, kb);
                #pragma unroll
                for (int nn = 0; nn < 2; ++nn)
                    bm[nn] = load_b(w_l2, HID, l15, half, kb, nt0 + nn);
                #pragma unroll
                for (int nn = 0; nn < 2; ++nn)
                    #pragma unroll
                    for (int mt = 0; mt < 2; ++mt)
                        acc[nn][mt] = __builtin_amdgcn_wmma_f32_16x16x32_bf16(
                            false, am[mt], false, bm[nn], (short)0, acc[nn][mt],
                            false, false);
            }
            #pragma unroll
            for (int nn = 0; nn < 2; ++nn) {
                const int col = (nt0 + nn) * 16 + l15;
                const float bv = bias2[mlp][col];
                const float wv = wr_s[mlp][col];
                #pragma unroll
                for (int mt = 0; mt < 2; ++mt)
                    #pragma unroll
                    for (int r = 0; r < 8; ++r)
                        racc[mt][r] += fmaxf(acc[nn][mt][r] + bv, 0.0f) * wv;
            }
        }

        // after last node of this MLP: reduce 16 columns/lane-group, add rows
        if (node == 2) {
            #pragma unroll
            for (int mt = 0; mt < 2; ++mt)
                #pragma unroll
                for (int r = 0; r < 8; ++r) {
                    float x = racc[mt][r];
                    x += __shfl_xor(x, 1, 32);
                    x += __shfl_xor(x, 2, 32);
                    x += __shfl_xor(x, 4, 32);
                    x += __shfl_xor(x, 8, 32);
                    if (l15 == 0)
                        atomicAdd(&qacc[mlp][mt * 16 + half * 8 + r], x);
                }
        }
        __syncthreads();
    }

    // ---- final: q = acc + br ----
    if (tid < 2 * MROWS) {
        int mlp = tid >> 5;
        int r   = tid & (MROWS - 1);
        int rg  = b0 + r;
        if (rg < Btot)
            out[(size_t)mlp * Btot + rg] = qacc[mlp][r] + (mlp ? br2[0] : br1[0]);
    }
}

extern "C" void kernel_launch(void* const* d_in, const int* in_sizes, int n_in,
                              void* d_out, int out_size, void* d_ws, size_t ws_size,
                              hipStream_t stream) {
    const float* obs      = (const float*)d_in[0];
    // d_in[1] = act (unused by the reference)
    const float* ef       = (const float*)d_in[2];
    const int*   edges_to = (const int*)d_in[3];
    const float* w1 = (const float*)d_in[4];
    const float* b1 = (const float*)d_in[5];
    const float* w2 = (const float*)d_in[6];
    const float* b2 = (const float*)d_in[7];
    const float* w3 = (const float*)d_in[8];
    const float* b3 = (const float*)d_in[9];
    const float* w4 = (const float*)d_in[10];
    const float* b4 = (const float*)d_in[11];
    const float* wr1 = (const float*)d_in[12];
    const float* br1 = (const float*)d_in[13];
    const float* wr2 = (const float*)d_in[14];
    const float* br2 = (const float*)d_in[15];

    u16_t* ws = (u16_t*)d_ws;
    const int Btot = in_sizes[0] / OBSD;

    const int convGrid = (WS_TOT + 255) / 256;
    conv_weights<<<convGrid, 256, 0, stream>>>(w1, w2, w3, w4, ws);

    const int grid = (Btot + MROWS - 1) / MROWS;
    gnn_critic_main<<<grid, 256, 0, stream>>>(
        obs, ef, edges_to, b1, b2, b3, b4, wr1, br1, wr2, br2,
        ws, (float*)d_out, Btot);
}
